// FeatureSharingLayer_8486855377278
// MI455X (gfx1250) — matrix operand order repaired and checked by reading, not compiled, not run on hardware
//
#include <hip/hip_runtime.h>
#include <hip/hip_bf16.h>

// ---------------------------------------------------------------------------
// FeatureSharingLayer, fully fused for MI455X (gfx1250, wave32, WMMA).
//   M = B*N_TASKS = 262144 rows, 6 chained 128x128 GEMMs per row.
//   Math: bf16 A/B fragments, f32 accumulate via v_wmma_f32_16x16x32_bf16.
//   concatA @ fuse_w1.T is collapsed to A @ W1eff.T (W1eff = sum of 8 chunks).
// ---------------------------------------------------------------------------

#define DIMK   128
#define WSTR   136   // padded bf16 elements per weight/activation LDS row
#define XSTR   132   // padded f32 elements per x0 LDS row
#define NBLOCKS 512

typedef __attribute__((ext_vector_type(16))) __bf16 v16bf;
typedef __attribute__((ext_vector_type(8)))  float  v8f;

union FragU { v16bf v; uint4 u[2]; };

__device__ __forceinline__ v8f bwmma(v16bf a, v16bf b, v8f c) {
  // 8 args: (neg_a, A, neg_b, B, c_mod, C, reuse_a, reuse_b)
  return __builtin_amdgcn_wmma_f32_16x16x32_bf16(false, a, false, b, (short)0, c,
                                                 false, false);
}

// ---------------------------------------------------------------------------
// Prep: build 6 bf16 weight matrices (row-major [n][k]) in workspace:
//   slot 0: Wv  = in_proj_w[256:384]     slot 1: out_w
//   slot 2: W1eff[j][d]=sum_k fuse_w1[j][k*128+d]
//   slot 3: fuse_w2   slot 4: upd_w1     slot 5: upd_w2
// ---------------------------------------------------------------------------
__global__ void fsl_prep_weights(const float* __restrict__ in_proj_w,
                                 const float* __restrict__ fuse_w1,
                                 const float* __restrict__ out_w,
                                 const float* __restrict__ fuse_w2,
                                 const float* __restrict__ upd_w1,
                                 const float* __restrict__ upd_w2,
                                 __bf16* __restrict__ ws) {
  int e   = blockIdx.x * 256 + threadIdx.x;   // 0 .. 6*16384-1
  int mi  = e >> 14;
  int idx = e & 16383;
  float v;
  if (mi == 0)      v = in_proj_w[2 * DIMK * DIMK + idx];
  else if (mi == 1) v = out_w[idx];
  else if (mi == 2) {
    int n = idx >> 7, d = idx & 127;
    float s = 0.f;
#pragma unroll
    for (int k = 0; k < 8; ++k) s += fuse_w1[n * 1024 + k * 128 + d];
    v = s;
  }
  else if (mi == 3) v = fuse_w2[idx];
  else if (mi == 4) v = upd_w1[idx];
  else              v = upd_w2[idx];
  ws[e] = (__bf16)v;
}

// ---------------------------------------------------------------------------
// One GEMM stage: out = [relu]( act(16-row frags) @ W.T + bias [+Ares] ) [+x0]
// Wave owns rows [rowBase, rowBase+32) x cols [colBase, colBase+64).
// ---------------------------------------------------------------------------
template <bool RELU, bool SAVE_A, bool ADD_A, bool FINAL>
__device__ __forceinline__ void do_stage(__bf16* act, const __bf16* W,
                                         const float* bias, v8f Ares[2][4],
                                         const float* x0, float* outg,
                                         int l15, int half, int rowBase,
                                         int colBase) {
  // --- A fragments: 2 m-tiles x 4 K-chunks, per ISA 16-bit A 16x32 layout ---
  FragU a[2][4];
#pragma unroll
  for (int m = 0; m < 2; ++m) {
    const __bf16* arow = act + (size_t)(rowBase + 16 * m + l15) * WSTR;
#pragma unroll
    for (int c = 0; c < 4; ++c) {
      const int k0 = 32 * c + 8 * half;
      a[m][c].u[0] = *(const uint4*)(arow + k0);
      a[m][c].u[1] = *(const uint4*)(arow + k0 + 16);
    }
  }
#pragma unroll
  for (int t = 0; t < 4; ++t) {
    const int n0 = colBase + 16 * t;
    v8f acc[2] = {};
#pragma unroll
    for (int c = 0; c < 4; ++c) {
      // B fragment: B[k][n] = W[n][k]; lane = col, k = 8*half+s / 16+8*half+(s-8).
      FragU b;
      const __bf16* wp = W + (size_t)(n0 + l15) * WSTR + 32 * c + 8 * half;
      b.u[0] = *(const uint4*)(wp);
      b.u[1] = *(const uint4*)(wp + 16);
      acc[0] = bwmma(a[0][c].v, b.v, acc[0]);   // B reused for both m-tiles
      acc[1] = bwmma(a[1][c].v, b.v, acc[1]);
    }
    const float bn = bias[n0 + l15];
#pragma unroll
    for (int m = 0; m < 2; ++m) {
#pragma unroll
      for (int r = 0; r < 8; ++r) {
        float v = acc[m][r] + bn;
        if (ADD_A) v += Ares[m][t][r];
        if (RELU) v = fmaxf(v, 0.0f);
        if (SAVE_A) Ares[m][t][r] = v;
        const int row = rowBase + 16 * m + 8 * half + r;   // C-layout row
        const int col = n0 + l15;                          // C-layout col
        if (FINAL) {
          outg[(size_t)row * DIMK + col] = v + x0[(size_t)row * XSTR + col];
        } else {
          act[(size_t)row * WSTR + col] = (__bf16)v;
        }
      }
    }
  }
}

// ---------------------------------------------------------------------------
// Main fused kernel: 256 threads (8 waves), 128 rows per block-iteration.
// LDS: 6x128xWSTR bf16 weights | 128xWSTR bf16 act | 128xXSTR f32 x0 | biases
// ---------------------------------------------------------------------------
__global__ __launch_bounds__(256, 1) void fsl_fused_chain(
    const float* __restrict__ tf, const __bf16* __restrict__ wsW,
    const float* __restrict__ in_proj_b, const float* __restrict__ out_b,
    const float* __restrict__ fuse_b1, const float* __restrict__ fuse_b2,
    const float* __restrict__ upd_b1, const float* __restrict__ upd_b2,
    float* __restrict__ outp, int nRowBlocks) {
  extern __shared__ unsigned char smem[];
  __bf16* ldsW = (__bf16*)smem;                       // 6*128*WSTR halves
  __bf16* act  = ldsW + 6 * 128 * WSTR;               // 128*WSTR halves
  float*  x0   = (float*)(act + 128 * WSTR);          // 128*XSTR floats
  float*  bias = x0 + 128 * XSTR;                     // 6*128 floats

  const int tid = threadIdx.x;

  // ---- cooperative weight load: global bf16 -> padded LDS rows ----
  for (int c = tid; c < 6 * 128 * 2; c += 256) {
    int mi = c >> 8;           // matrix 0..5
    int row = (c >> 1) & 127;  // row in matrix
    int hh = c & 1;            // 128-byte half of the row
    const uint4* src = (const uint4*)(wsW + mi * 16384 + row * 128 + hh * 64);
    uint4* dst = (uint4*)(ldsW + (size_t)mi * 128 * WSTR + (size_t)row * WSTR + hh * 64);
#pragma unroll
    for (int i = 0; i < 8; ++i) dst[i] = src[i];
  }
  if (tid < 128) {
    bias[0 * 128 + tid] = in_proj_b[256 + tid];
    bias[1 * 128 + tid] = out_b[tid];
    bias[2 * 128 + tid] = fuse_b1[tid];
    bias[3 * 128 + tid] = fuse_b2[tid];
    bias[4 * 128 + tid] = upd_b1[tid];
    bias[5 * 128 + tid] = upd_b2[tid];
  }
  __syncthreads();

  const int lane = tid & 31;
  const int l15  = lane & 15;
  const int half = lane >> 4;
  const int wave = tid >> 5;
  const int rowBase = (wave >> 1) * 32;   // 4 row-groups of 32 rows
  const int colBase = (wave & 1) * 64;    // 2 col-groups of 64 cols

  for (int blk = blockIdx.x; blk < nRowBlocks; blk += gridDim.x) {
    const size_t gRow = (size_t)blk * 128;

    // ---- load X0 tile: f32 -> x0 LDS (residual), bf16 -> act LDS ----
    {
      const int row  = tid >> 1;
      const int col0 = (tid & 1) * 64;
      const float4* src = (const float4*)(tf + (gRow + row) * DIMK + col0);
      float4* xd = (float4*)(x0 + (size_t)row * XSTR + col0);
      __bf16* ad = act + (size_t)row * WSTR + col0;
#pragma unroll
      for (int i = 0; i < 16; ++i) {
        float4 f = src[i];
        xd[i] = f;
        ad[4 * i + 0] = (__bf16)f.x;
        ad[4 * i + 1] = (__bf16)f.y;
        ad[4 * i + 2] = (__bf16)f.z;
        ad[4 * i + 3] = (__bf16)f.w;
      }
    }
    __syncthreads();

    v8f Ares[2][4];
    float* og = outp + gRow * DIMK;

    // S1: T1 = X0 @ Wv.T + bv
    do_stage<false, false, false, false>(act, ldsW + 0 * 128 * WSTR, bias + 0 * 128,
                                         Ares, x0, og, l15, half, rowBase, colBase);
    __syncthreads();
    // S2: A = T1 @ out_w.T + out_b   (saved as residual)
    do_stage<false, true, false, false>(act, ldsW + 1 * 128 * WSTR, bias + 1 * 128,
                                        Ares, x0, og, l15, half, rowBase, colBase);
    __syncthreads();
    // S3: H = relu(A @ W1eff.T + fuse_b1)
    do_stage<true, false, false, false>(act, ldsW + 2 * 128 * WSTR, bias + 2 * 128,
                                        Ares, x0, og, l15, half, rowBase, colBase);
    __syncthreads();
    // S4: Ahat = H @ fuse_w2.T + fuse_b2 + A
    do_stage<false, false, true, false>(act, ldsW + 3 * 128 * WSTR, bias + 3 * 128,
                                        Ares, x0, og, l15, half, rowBase, colBase);
    __syncthreads();
    // S5: H2 = relu(Ahat @ upd_w1.T + upd_b1)
    do_stage<true, false, false, false>(act, ldsW + 4 * 128 * WSTR, bias + 4 * 128,
                                        Ares, x0, og, l15, half, rowBase, colBase);
    __syncthreads();
    // S6: out = H2 @ upd_w2.T + upd_b2 + X0   (write f32 to global)
    do_stage<false, false, false, true>(act, ldsW + 5 * 128 * WSTR, bias + 5 * 128,
                                        Ares, x0, og, l15, half, rowBase, colBase);
    __syncthreads();   // WAR: next iteration overwrites act/x0
  }
}

// ---------------------------------------------------------------------------
extern "C" void kernel_launch(void* const* d_in, const int* in_sizes, int n_in,
                              void* d_out, int out_size, void* d_ws, size_t ws_size,
                              hipStream_t stream) {
  const float* tf        = (const float*)d_in[0];
  const float* in_proj_w = (const float*)d_in[1];
  const float* in_proj_b = (const float*)d_in[2];
  const float* out_w     = (const float*)d_in[3];
  const float* out_b     = (const float*)d_in[4];
  const float* fuse_w1   = (const float*)d_in[5];
  const float* fuse_b1   = (const float*)d_in[6];
  const float* fuse_w2   = (const float*)d_in[7];
  const float* fuse_b2   = (const float*)d_in[8];
  const float* upd_w1    = (const float*)d_in[9];
  const float* upd_b1    = (const float*)d_in[10];
  const float* upd_w2    = (const float*)d_in[11];
  const float* upd_b2    = (const float*)d_in[12];
  float*  outp = (float*)d_out;
  __bf16* ws   = (__bf16*)d_ws;   // 6 * 128*128 bf16 = 192 KB

  // Build bf16 weights (incl. W1eff fold) in workspace.
  fsl_prep_weights<<<6 * 16384 / 256, 256, 0, stream>>>(in_proj_w, fuse_w1, out_w,
                                                        fuse_w2, upd_w1, upd_w2, ws);

  const int M = in_sizes[0] / DIMK;       // 262144 rows
  const int nRowBlocks = M / 128;         // 2048
  const int grid = nRowBlocks < NBLOCKS ? nRowBlocks : NBLOCKS;

  const size_t shmem = (size_t)(6 * 128 * WSTR + 128 * WSTR) * sizeof(__bf16) +
                       (size_t)(128 * XSTR + 6 * 128) * sizeof(float);  // 314368 B
  (void)hipFuncSetAttribute((const void*)fsl_fused_chain,
                            hipFuncAttributeMaxDynamicSharedMemorySize, (int)shmem);
  fsl_fused_chain<<<grid, 256, shmem, stream>>>(tf, ws, in_proj_b, out_b, fuse_b1,
                                                fuse_b2, upd_b1, upd_b2, outp,
                                                nRowBlocks);
}
